// ScannedTransformerHypernetwork_48430051230162
// MI455X (gfx1250) — matrix operand order repaired and checked by reading, not compiled, not run on hardware
//
#include <hip/hip_runtime.h>
#include <hip/hip_bf16.h>

// ---------------------------------------------------------------------------
// Scanned transformer hypernetwork for MI455X (gfx1250, wave32, WMMA).
// - All large GEMMs: v_wmma_f32_16x16x32_bf16, f32 accumulate.
// - Weights repacked once per launch into WMMA-fragment-tiled bf16 (25MB,
//   L2-resident across the 32 sequential scan steps).
// - Activations carry a bf16 shadow copy so GEMM A-tiles are staged into LDS
//   with global_load_async_to_lds_b128 (ASYNCcnt) -- no per-tile conversion.
// - Tile coverage is exact per launch config: NO divergent guards in the hot
//   loop (round-2 guards caused full accumulator spills around each WMMA).
// ---------------------------------------------------------------------------

typedef __attribute__((ext_vector_type(16))) __bf16         v16bf;
typedef __attribute__((ext_vector_type(16))) unsigned short v16us;
typedef __attribute__((ext_vector_type(8)))  float          v8f;

#define T_STEPS 32
#define NAG     16        // agents
#define LAG     33        // 1 + LA tokens per agent
#define RA      (NAG*LAG) // 528 agent rows (multiple of 16)
#define LMX     49        // 1 + LM + NA mixer tokens
#define RM      64        // padded mixer rows
#define HDIM    512
#define NHEAD   8
#define HD      64
#define FFD     2048
#define AOUT    16

// fp32 -> bf16 (round to nearest even)
__device__ __forceinline__ unsigned short bf_bits(float f) {
  unsigned u = __builtin_bit_cast(unsigned, f);
  u += 0x7FFFu + ((u >> 16) & 1u);
  return (unsigned short)(u >> 16);
}

// ---------------------------------------------------------------------------
// Weight repack: row-major fp32 [K x N] -> fragment-tiled bf16.
// B-fragment layout (16-bit, 32x16 per k-chunk):
//   lane = n_local(0..15) | half<<4 ; element e(0..15): k_local = half*16 + e
// Linear index: (((kchunk*Ntiles + ntile)*32 + lane)*16 + e)
// ---------------------------------------------------------------------------
__global__ void conv_weight(const float* __restrict__ W,
                            unsigned short* __restrict__ out, int K, int N) {
  int t = blockIdx.x * blockDim.x + threadIdx.x;
  if (t >= K * N) return;
  int e      = t & 15;
  int lane   = (t >> 4) & 31;
  int tile   = t >> 9;
  int ntiles = N >> 4;
  int ntile  = tile % ntiles;
  int ktile  = tile / ntiles;
  int k = (ktile << 5) + ((lane >> 4) << 4) + e;
  int n = (ntile << 4) + (lane & 15);
  out[t] = bf_bits(W[(size_t)k * N + n]);
}

// ---------------------------------------------------------------------------
// GEMM: [M x N] = Xbf[M x K](bf16) * Wfrag(bf16 tiled) + bias; opt ReLU.
// Outputs: Y (fp32, nullable) and Ybf (bf16 shadow, nullable).
// NT = n-tiles per wave (compile-time). Launch guarantees exact coverage:
//   big GEMMs: 128 thr (4 waves) x NT=4, grid.x = ntiles/16  (N=512 or 2048)
//   q_vals  :  32 thr (1 wave)  x NT=1, grid.x = 1           (N=16)
// A tile (16x64 bf16, 2KB) staged via async-LDS copy; 2*NT WMMAs per stage.
// A fragment uses contiguous k order, same as the B repack:
//   lane: m=lane&15, half=lane>>4 ; vgpr v: K0=2v+16*half
// ---------------------------------------------------------------------------
template <int NT>
__global__ void gemm_bias(const unsigned short* __restrict__ Xbf,
                          const unsigned short* __restrict__ Wf,
                          const float* __restrict__ bias,
                          float* __restrict__ Y,
                          unsigned short* __restrict__ Ybf,
                          int N, int K, int relu) {
  __shared__ unsigned short Atile[16 * 64];
  const int tid    = threadIdx.x;
  const int lane   = tid & 31;
  const int wid    = tid >> 5;
  const int nwaves = blockDim.x >> 5;
  const int ntiles = N >> 4;
  const int m0     = blockIdx.y << 4;
  const int nt0    = (blockIdx.x * nwaves + wid) * NT;

  v8f acc[NT];
#pragma unroll
  for (int j = 0; j < NT; ++j) acc[j] = v8f{0.f,0.f,0.f,0.f,0.f,0.f,0.f,0.f};

  const int m = lane & 15, half = lane >> 4;
  const unsigned ldsBase = (unsigned)(size_t)(&Atile[0]);

  const int kIter = K >> 6;
  for (int kt = 0; kt < kIter; ++kt) {
    // async stage: copy the 16x64 bf16 tile (128 x 16B chunks) into LDS
    for (int ii = tid; ii < 128; ii += blockDim.x) {
      int ar = ii >> 3;              // row 0..15
      int ac = (ii & 7) * 8;         // col (ushorts)
      unsigned long long ga = (unsigned long long)(size_t)(
          Xbf + (size_t)(m0 + ar) * K + (kt << 6) + ac);
      unsigned ldsW = ldsBase + (unsigned)ii * 16u;
      asm volatile("global_load_async_to_lds_b128 %0, %1, off"
                   :: "v"(ldsW), "v"(ga) : "memory");
    }
    asm volatile("s_wait_asynccnt 0x0" ::: "memory");
    __syncthreads();

    // prefetch next k-step's first B fragment for this wave (L2 temporal)
    const unsigned short* pf =
        Wf + (((size_t)((kt + 1) << 1) * ntiles + nt0) << 5) * 16;
    __builtin_prefetch(pf, 0, 1);

#pragma unroll
    for (int kk = 0; kk < 2; ++kk) {
      v16us aus;
#pragma unroll
      for (int v = 0; v < 8; ++v) {
        int k0 = 2 * v + 16 * half;
        unsigned pair = *(const unsigned*)&Atile[m * 64 + kk * 32 + k0];
        aus[2 * v]     = (unsigned short)(pair & 0xFFFFu);
        aus[2 * v + 1] = (unsigned short)(pair >> 16);
      }
      v16bf a = __builtin_bit_cast(v16bf, aus);
      int kc = (kt << 1) | kk;               // 32-wide k-chunk index
#pragma unroll
      for (int j = 0; j < NT; ++j) {         // straight-line: no guards
        const v16us* bp = (const v16us*)(Wf +
            ((((size_t)kc * ntiles + (nt0 + j)) << 5) + (size_t)lane) * 16);
        v16bf b = __builtin_bit_cast(v16bf, *bp);
        acc[j] = __builtin_amdgcn_wmma_f32_16x16x32_bf16(
            false, a, false, b, (short)0, acc[j], false, false);
      }
    }
    __syncthreads();
  }

  const int rbase = m0 + (half << 3);
#pragma unroll
  for (int j = 0; j < NT; ++j) {
    int col  = ((nt0 + j) << 4) + m;
    float bv = bias[col];
#pragma unroll
    for (int v = 0; v < 8; ++v) {            // D layout: row rbase+v, col
      float val = acc[j][v] + bv;
      if (relu) val = fmaxf(val, 0.f);
      if (Y)   Y  [(size_t)(rbase + v) * N + col] = val;
      if (Ybf) Ybf[(size_t)(rbase + v) * N + col] = bf_bits(val);
    }
  }
}

// ---------------------------------------------------------------------------
// FAVOR linear attention, one block per (batch, head). L <= 49, HD = 64.
// o = (qp @ (kp^T v)) / (qp @ sum_l kp),  qp/kp = relu(.)+1e-3.  bf16 out.
// ---------------------------------------------------------------------------
__global__ void favor_kernel(const float* __restrict__ Q,
                             const float* __restrict__ Km,
                             const float* __restrict__ V,
                             unsigned short* __restrict__ O, int L) {
  __shared__ float kp[49 * 64];
  __shared__ float vv[49 * 64];
  __shared__ float kv[64 * 64];
  __shared__ float ksum[64];
  __shared__ float den[49];
  const int b = blockIdx.x >> 3;
  const int h = blockIdx.x & 7;
  const int base = b * L;
  const int c0 = h * HD;
  const int tid = threadIdx.x;

  for (int i = tid; i < L * 64; i += blockDim.x) {
    int l = i >> 6, d = i & 63;
    size_t gi = (size_t)(base + l) * HDIM + c0 + d;
    kp[i] = fmaxf(Km[gi], 0.f) + 1e-3f;
    vv[i] = V[gi];
  }
  __syncthreads();
  if (tid < 64) {
    float s = 0.f;
    for (int l = 0; l < L; ++l) s += kp[l * 64 + tid];
    ksum[tid] = s;
  }
  for (int i = tid; i < 4096; i += blockDim.x) {
    int mm = i >> 6, d = i & 63;
    float s = 0.f;
    for (int l = 0; l < L; ++l) s += kp[l * 64 + mm] * vv[l * 64 + d];
    kv[i] = s;
  }
  __syncthreads();
  for (int i = tid; i < L * 64; i += blockDim.x) {    // reuse kp[] as qp
    int l = i >> 6, d = i & 63;
    kp[i] = fmaxf(Q[(size_t)(base + l) * HDIM + c0 + d], 0.f) + 1e-3f;
  }
  __syncthreads();
  if (tid < L) {
    float s = 0.f;
    for (int mm = 0; mm < 64; ++mm) s += kp[tid * 64 + mm] * ksum[mm];
    den[tid] = s;
  }
  __syncthreads();
  for (int i = tid; i < L * 64; i += blockDim.x) {
    int l = i >> 6, d = i & 63;
    float s = 0.f;
    for (int mm = 0; mm < 64; ++mm) s += kp[l * 64 + mm] * kv[mm * 64 + d];
    O[(size_t)(base + l) * HDIM + c0 + d] = bf_bits(s / den[l]);
  }
}

// ---------------------------------------------------------------------------
// LayerNorm(2*y)*g + b over H=512, one wave32 per row; fp32 + bf16 outputs.
// ---------------------------------------------------------------------------
__global__ void ln2_kernel(const float* __restrict__ Y,
                           const float* __restrict__ g,
                           const float* __restrict__ b,
                           float* __restrict__ Xo,
                           unsigned short* __restrict__ Xbf, int rows) {
  int lane = threadIdx.x & 31;
  int wid  = threadIdx.x >> 5;
  int row  = blockIdx.x * (blockDim.x >> 5) + wid;
  if (row >= rows) return;
  const float* yr = Y + (size_t)row * HDIM;
  float vals[16];
  float s = 0.f;
#pragma unroll
  for (int i = 0; i < 16; ++i) { vals[i] = 2.f * yr[lane + 32 * i]; s += vals[i]; }
#pragma unroll
  for (int off = 16; off >= 1; off >>= 1) s += __shfl_xor(s, off, 32);
  float mu = s * (1.f / 512.f);
  float vs = 0.f;
#pragma unroll
  for (int i = 0; i < 16; ++i) { float d = vals[i] - mu; vs += d * d; }
#pragma unroll
  for (int off = 16; off >= 1; off >>= 1) vs += __shfl_xor(vs, off, 32);
  float inv = rsqrtf(vs * (1.f / 512.f) + 1e-6f);
  float* xr = Xo + (size_t)row * HDIM;
  unsigned short* xb = Xbf + (size_t)row * HDIM;
#pragma unroll
  for (int i = 0; i < 16; ++i) {
    int c = lane + 32 * i;
    float o = (vals[i] - mu) * inv * g[c] + b[c];
    xr[c] = o;
    xb[c] = bf_bits(o);
  }
}

// ---------------------------------------------------------------------------
// Step-assembly kernels (fp32 + bf16 shadows where GEMMs consume them)
// ---------------------------------------------------------------------------
__global__ void build_agent_x(float* __restrict__ XA,
                              unsigned short* __restrict__ XAbf,
                              const float* __restrict__ hsA,
                              const int* __restrict__ done_t,
                              const float* __restrict__ ae_t) {
  int i = blockIdx.x * blockDim.x + threadIdx.x;     // RA*H
  if (i >= RA * HDIM) return;
  int r = i >> 9, c = i & 511;
  int b = r / LAG, l = r - b * LAG;
  float v;
  if (l == 0) v = done_t[b] ? 0.f : hsA[b * HDIM + c];
  else        v = ae_t[(size_t)(b * 32 + (l - 1)) * HDIM + c];
  XA[i] = v;
  XAbf[i] = bf_bits(v);
}

__global__ void build_mixer_x(float* __restrict__ XM,
                              unsigned short* __restrict__ XMbf,
                              const float* __restrict__ hsM,
                              const int* __restrict__ dg_t,
                              const float* __restrict__ me_t,
                              const float* __restrict__ hsA) {
  int i = blockIdx.x * blockDim.x + threadIdx.x;     // RM*H
  if (i >= RM * HDIM) return;
  int r = i >> 9, c = i & 511;
  float v;
  if (r == 0)        v = dg_t[0] ? 0.f : hsM[c];
  else if (r <= 32)  v = me_t[(size_t)(r - 1) * HDIM + c];
  else if (r <= 48)  v = hsA[(size_t)(r - 33) * HDIM + c];
  else               v = 0.f;                        // pad rows
  XM[i] = v;
  XMbf[i] = bf_bits(v);
}

__global__ void extract_hs_agent(float* __restrict__ hsA,
                                 unsigned short* __restrict__ hsAbf,
                                 const float* __restrict__ XA) {
  int i = blockIdx.x * blockDim.x + threadIdx.x;     // NAG*H
  if (i >= NAG * HDIM) return;
  int b = i >> 9, c = i & 511;
  float v = XA[(size_t)(b * LAG) * HDIM + c];
  hsA[i] = v;
  hsAbf[i] = bf_bits(v);
}

__global__ void finish_mixer(float* __restrict__ embOut,
                             float* __restrict__ hsM,
                             const float* __restrict__ XM) {
  int i = blockIdx.x * blockDim.x + threadIdx.x;     // LMX*H
  if (i >= LMX * HDIM) return;
  embOut[i] = XM[i];
  if (i < HDIM) hsM[i] = XM[i];
}

// ---------------------------------------------------------------------------
// Host-side orchestration
// ---------------------------------------------------------------------------
static inline void launch_gemm(const unsigned short* Xbf,
                               const unsigned short* Wf, const float* bias,
                               float* Y, unsigned short* Ybf,
                               int M, int N, int K, int relu, hipStream_t s) {
  int ntiles = N / 16;
  if (ntiles >= 16) {            // N = 512 or 2048: exact 4-wave x 4-tile
    dim3 grid(ntiles / 16, M / 16);
    gemm_bias<4><<<grid, 128, 0, s>>>(Xbf, Wf, bias, Y, Ybf, N, K, relu);
  } else {                       // q_vals: N = 16, single wave, single tile
    dim3 grid(1, M / 16);
    gemm_bias<1><<<grid, 32, 0, s>>>(Xbf, Wf, bias, Y, Ybf, N, K, relu);
  }
}

// bf16 weight region offsets (ushort elements) within one block
#define OFW_Q  0u
#define OFW_K  262144u
#define OFW_V  524288u
#define OFW_O  786432u
#define OFW_1  1048576u
#define OFW_2  2097152u
#define WBLK   3145728u   // ushorts per encoder block
#define OFW_QW (4u * WBLK)          // 12,582,912
#define WEND   (OFW_QW + 8192u)     // 12,591,104 ushorts

// bf16 activation shadow offsets (ushorts, after weights)
#define OFB_XA  (WEND + 0u)
#define OFB_ATA (WEND + 270336u)
#define OFB_HA  (WEND + 540672u)
#define OFB_XM  (WEND + 1622016u)
#define OFB_ATM (WEND + 1654784u)
#define OFB_HM  (WEND + 1687552u)
#define OFB_HSA (WEND + 1818624u)
#define USHORT_TOTAL (WEND + 1826816u)           // 14,417,920 ushorts

// fp32 scratch offsets (float elements) after the ushort region
#define FLOAT_BYTE_OFF ((size_t)USHORT_TOTAL * 2u)   // 28,835,840 B (16B align)
#define OFF_XA  0u
#define OFF_QA  270336u
#define OFF_KA  540672u
#define OFF_VA  811008u
#define OFF_YA  1081344u
#define OFF_XM  1351680u
#define OFF_QM  1384448u
#define OFF_KM  1417216u
#define OFF_VM  1449984u
#define OFF_YM  1482752u
#define OFF_HSA 1515520u
#define OFF_HSM 1523712u

static void run_block(float* X, unsigned short* Xbf, float* Qb, float* Kb,
                      float* Vb, unsigned short* Abf, float* Yb,
                      unsigned short* Hbf, int R, int L, int B,
                      const float* const* P, const unsigned short* W,
                      hipStream_t s) {
  // leaf order (alphabetical pytree): 0 b1, 1 b2, 2 bk, 3 bo, 4 bq, 5 bv,
  // 6 ln1_b, 7 ln1_g, 8 ln2_b, 9 ln2_g, 10 w1, 11 w2, 12 wk, 13 wo, 14 wq, 15 wv
  launch_gemm(Xbf, W + OFW_Q, P[4], Qb, nullptr, R, HDIM, HDIM, 0, s);
  launch_gemm(Xbf, W + OFW_K, P[2], Kb, nullptr, R, HDIM, HDIM, 0, s);
  launch_gemm(Xbf, W + OFW_V, P[5], Vb, nullptr, R, HDIM, HDIM, 0, s);
  favor_kernel<<<B * NHEAD, 256, 0, s>>>(Qb, Kb, Vb, Abf, L);
  launch_gemm(Abf, W + OFW_O, P[3], Yb, nullptr, R, HDIM, HDIM, 0, s);
  ln2_kernel<<<R / 8, 256, 0, s>>>(Yb, P[7], P[6], X, Xbf, R);
  launch_gemm(Xbf, W + OFW_1, P[0], nullptr, Hbf, R, FFD, HDIM, 1, s);
  launch_gemm(Hbf, W + OFW_2, P[1], Yb, nullptr, R, HDIM, FFD, 0, s);
  ln2_kernel<<<R / 8, 256, 0, s>>>(Yb, P[9], P[8], X, Xbf, R);
}

extern "C" void kernel_launch(void* const* d_in, const int* in_sizes, int n_in,
                              void* d_out, int out_size, void* d_ws,
                              size_t ws_size, hipStream_t stream) {
  (void)in_sizes; (void)n_in; (void)out_size; (void)ws_size;
  // Input order (pytree alphabetical): agent_embedding, mixer_embedding, done,
  // done_global, hs_agent0, hs_mixer0, then params leaves:
  //   agent_blocks[0..1] (16 leaves each), hyper_blocks[0..1], q_b, q_w
  const float* ae   = (const float*)d_in[0];   // (T, NA, LA, H)
  const float* me   = (const float*)d_in[1];   // (T, 1, LM, H)
  const int*   done = (const int*)d_in[2];     // (T, NA)
  const int*   dgl  = (const int*)d_in[3];     // (T, 1)
  const float* P[4][16];
  for (int blk = 0; blk < 4; ++blk)
    for (int j = 0; j < 16; ++j)
      P[blk][j] = (const float*)d_in[6 + blk * 16 + j];
  const float* q_b = (const float*)d_in[70];
  const float* q_w = (const float*)d_in[71];

  unsigned short* Wb = (unsigned short*)d_ws;
  float* F = (float*)((char*)d_ws + FLOAT_BYTE_OFF);
  float* XA  = F + OFF_XA;  float* QA  = F + OFF_QA;  float* KA  = F + OFF_KA;
  float* VA  = F + OFF_VA;  float* YA  = F + OFF_YA;
  float* XM  = F + OFF_XM;  float* QM  = F + OFF_QM;  float* KM  = F + OFF_KM;
  float* VM  = F + OFF_VM;  float* YM  = F + OFF_YM;
  float* hsA = F + OFF_HSA; float* hsM = F + OFF_HSM;
  unsigned short* XAbf  = Wb + OFB_XA;  unsigned short* ATAbf = Wb + OFB_ATA;
  unsigned short* HAbf  = Wb + OFB_HA;  unsigned short* XMbf  = Wb + OFB_XM;
  unsigned short* ATMbf = Wb + OFB_ATM; unsigned short* HMbf  = Wb + OFB_HM;
  unsigned short* hsAbf = Wb + OFB_HSA;
  float* out = (float*)d_out;

  // ---- one-time (per launch) weight repack to fragment-tiled bf16 ----
  for (int blk = 0; blk < 4; ++blk) {
    unsigned short* W = Wb + (size_t)blk * WBLK;
    conv_weight<<<(HDIM * HDIM) / 256, 256, 0, stream>>>(P[blk][14], W + OFW_Q, HDIM, HDIM);
    conv_weight<<<(HDIM * HDIM) / 256, 256, 0, stream>>>(P[blk][12], W + OFW_K, HDIM, HDIM);
    conv_weight<<<(HDIM * HDIM) / 256, 256, 0, stream>>>(P[blk][15], W + OFW_V, HDIM, HDIM);
    conv_weight<<<(HDIM * HDIM) / 256, 256, 0, stream>>>(P[blk][13], W + OFW_O, HDIM, HDIM);
    conv_weight<<<(HDIM * FFD)  / 256, 256, 0, stream>>>(P[blk][10], W + OFW_1, HDIM, FFD);
    conv_weight<<<(FFD * HDIM)  / 256, 256, 0, stream>>>(P[blk][11], W + OFW_2, FFD, HDIM);
  }
  conv_weight<<<(HDIM * AOUT) / 256, 256, 0, stream>>>(q_w, Wb + OFW_QW, HDIM, AOUT);

  // ---- initial hidden states ----
  hipMemcpyAsync(hsA, d_in[4], NAG * HDIM * sizeof(float),
                 hipMemcpyDeviceToDevice, stream);
  hipMemcpyAsync(hsM, d_in[5], HDIM * sizeof(float),
                 hipMemcpyDeviceToDevice, stream);

  // ---- sequential scan over T steps ----
  for (int t = 0; t < T_STEPS; ++t) {
    build_agent_x<<<(RA * HDIM) / 256, 256, 0, stream>>>(
        XA, XAbf, hsA, done + t * NAG, ae + (size_t)t * NAG * 32 * HDIM);
    for (int blk = 0; blk < 2; ++blk)
      run_block(XA, XAbf, QA, KA, VA, ATAbf, YA, HAbf, RA, LAG, NAG,
                P[blk], Wb + (size_t)blk * WBLK, stream);
    extract_hs_agent<<<(NAG * HDIM) / 256, 256, 0, stream>>>(hsA, hsAbf, XA);
    // q_vals: (16 x 512) @ (512 x 16) + q_b, straight into d_out
    launch_gemm(hsAbf, Wb + OFW_QW, q_b, out + (size_t)t * NAG * AOUT, nullptr,
                NAG, AOUT, HDIM, 0, stream);
    build_mixer_x<<<(RM * HDIM) / 256, 256, 0, stream>>>(
        XM, XMbf, hsM, dgl + t, me + (size_t)t * 32 * HDIM, hsA);
    for (int blk = 2; blk < 4; ++blk)
      run_block(XM, XMbf, QM, KM, VM, ATMbf, YM, HMbf, RM, LMX, 1,
                P[blk], Wb + (size_t)blk * WBLK, stream);
    finish_mixer<<<(LMX * HDIM + 255) / 256, 256, 0, stream>>>(
        out + (size_t)T_STEPS * NAG * AOUT + (size_t)t * LMX * HDIM, hsM, XM);
  }
}
